// PhraseAttentionExtractor_34265249088252
// MI455X (gfx1250) — matrix operand order repaired and checked by reading, not compiled, not run on hardware
//
#include <hip/hip_runtime.h>
#include <hip/hip_bf16.h>
#include <math.h>

#define Bn 8
#define Ln 512
#define Hn 768
#define Pn 5
#define Mn 32
#define Dn 256
#define LP (Ln * Pn)        // 2560
#define ROWS (Bn * LP)      // 20480
#define KF (4 * Hn)         // 3072
#define KSTEPS (KF / 32)    // 96
#define TPW 6               // N-tiles per wave (8 waves * 6 = 48 = 768/16)
#define RT 2                // row tiles (16 rows each) per block -> 32 rows
#define Gn (Hn / 2)         // 384

typedef __attribute__((ext_vector_type(16))) __bf16 v16bf;
typedef __attribute__((ext_vector_type(8)))  float  v8f;

union Frag {
    v16bf v;
    uint4 q[2];
};

__device__ __forceinline__ unsigned short f2bf(float x) {
    unsigned int u = __float_as_uint(x);
    unsigned int r = u + 0x7FFFu + ((u >> 16) & 1u); // round-to-nearest-even
    return (unsigned short)(r >> 16);
}

struct RowP {
    const float* hs;   // hidden[b, l, :]
    const float* he;   // hidden[b, jc, :]
    const float* c0;   // csh[b, l, :]
    const float* c1;   // csh[b, jc+1, :]
    float il;          // 1/(p+1)
};

__device__ __forceinline__ RowP row_params(int R, const float* hidden, const float* csh) {
    RowP rp;
    int b = R / LP; int rem = R - b * LP;
    int l = rem / Pn; int p = rem - l * Pn;
    int jc = l + p; if (jc > Ln - 1) jc = Ln - 1;
    rp.hs = hidden + ((size_t)(b * Ln + l)) * Hn;
    rp.he = hidden + ((size_t)(b * Ln + jc)) * Hn;
    rp.c0 = csh + ((size_t)(b * (Ln + 1) + l)) * Hn;
    rp.c1 = csh + ((size_t)(b * (Ln + 1) + jc + 1)) * Hn;
    rp.il = 1.f / (float)(p + 1);
    return rp;
}

__device__ __forceinline__ float feat_val(int seg, int c, const RowP& r) {
    if (seg == 0) return r.hs[c];                     // start
    if (seg == 1) return r.he[c];                     // end
    if (seg == 2) return (r.c1[c] - r.c0[c]) * r.il;  // mean over span
    return r.he[c] - r.hs[c];                         // end - start
}

// ---------------------------------------------------------------------------
// K0: per-batch prefix sums of hidden (csh) and mask (csm)
// ---------------------------------------------------------------------------
__global__ __launch_bounds__(256) void k_cumsum(const float* __restrict__ hidden,
                                                const int* __restrict__ amask,
                                                float* __restrict__ csh,
                                                int* __restrict__ csm) {
    int b = blockIdx.x;
    for (int c = threadIdx.x; c < Hn; c += 256) {
        float acc = 0.f;
        csh[((size_t)b * (Ln + 1)) * Hn + c] = 0.f;
        const float* hp = hidden + ((size_t)b * Ln) * Hn + c;
        float* cp = csh + ((size_t)(b * (Ln + 1) + 1)) * Hn + c;
        for (int l = 0; l < Ln; ++l) {
            acc += hp[(size_t)l * Hn];
            cp[(size_t)l * Hn] = acc;
        }
    }
    if (threadIdx.x == 0) {
        int acc = 0;
        csm[b * (Ln + 1)] = 0;
        for (int l = 0; l < Ln; ++l) {
            acc += amask[b * Ln + l];
            csm[b * (Ln + 1) + l + 1] = acc;
        }
    }
}

// ---------------------------------------------------------------------------
// K1: W1 (3072x768 f32, row-major) -> w1t (768x3072 bf16, row = output col)
// ---------------------------------------------------------------------------
__global__ __launch_bounds__(256) void k_prep_w1(const float* __restrict__ W1,
                                                 unsigned short* __restrict__ w1t) {
    int i = blockIdx.x * 256 + threadIdx.x;
    if (i >= Hn * KF) return;
    int n = i / KF;
    int f = i - n * KF;
    w1t[i] = f2bf(W1[(size_t)f * Hn + n]);
}

// ---------------------------------------------------------------------------
// K2: main WMMA GEMM: scores = mask(relu(feat @ W1 + b1) @ w2 + b2)
//     One block = 32 rows (two 16-row A tiles). 8 waves x 6 N-tiles = 768 cols.
//     Each B fragment feeds 2 WMMAs (rows 0-15 and 16-31).
// ---------------------------------------------------------------------------
__global__ __launch_bounds__(256) void k_score(const float* __restrict__ hidden,
                                               const float* __restrict__ csh,
                                               const int* __restrict__ csm,
                                               const unsigned short* __restrict__ w1t,
                                               const float* __restrict__ b1,
                                               const float* __restrict__ w2,
                                               const float* __restrict__ b2,
                                               float* __restrict__ scores) {
    __shared__ __align__(16) unsigned short a_lds[32 * 32];
    __shared__ float sc[32];

    const int tid = threadIdx.x;
    const int lane = tid & 31;
    const int wave = tid >> 5;           // 0..7
    const int rowBase = blockIdx.x * 32;
    const int kk = lane;                 // K position staged by this thread

    // --- per-thread row params: this thread stages rows wave + {0,8,16,24} ---
    RowP rp0 = row_params(rowBase + wave, hidden, csh);
    RowP rp1 = row_params(rowBase + wave + 8, hidden, csh);
    RowP rp2 = row_params(rowBase + wave + 16, hidden, csh);
    RowP rp3 = row_params(rowBase + wave + 24, hidden, csh);

    // --- per-wave WMMA fragment constants ---
    const int nn = lane & 15;            // N within tile / M within A tile
    const int hi = lane >> 4;            // lane half
    const int kofsA = hi * 8;            // A: lanes<16 -> K 0..7/16..23; >=16 -> 8..15/24..31
    const int khalfB = hi * 8;           // B: lanes<16 -> K 0..7/16..23; >=16 -> 8..15/24..31
    const int ct0 = wave * TPW;

    float b1c[TPW], w2c[TPW];
#pragma unroll
    for (int t = 0; t < TPW; ++t) {
        int col = (ct0 + t) * 16 + nn;
        b1c[t] = b1[col];
        w2c[t] = w2[col];
    }

    v8f zero = {};
    v8f acc[RT][TPW];
#pragma unroll
    for (int q = 0; q < RT; ++q)
#pragma unroll
        for (int t = 0; t < TPW; ++t) acc[q][t] = zero;

    for (int kt = 0; kt < KSTEPS; ++kt) {
        // seg/c are pure functions of the scalar loop index (Hn = 24*32)
        const int seg = kt / 24;
        const int c = (kt - seg * 24) * 32 + kk;

        // stage A tile (32 rows x 32 K) in bf16
        a_lds[(wave)      * 32 + kk] = f2bf(feat_val(seg, c, rp0));
        a_lds[(wave + 8)  * 32 + kk] = f2bf(feat_val(seg, c, rp1));
        a_lds[(wave + 16) * 32 + kk] = f2bf(feat_val(seg, c, rp2));
        a_lds[(wave + 24) * 32 + kk] = f2bf(feat_val(seg, c, rp3));
        __syncthreads();

        Frag a0, a1;
        a0.q[0] = *(const uint4*)(&a_lds[nn * 32 + kofsA]);
        a0.q[1] = *(const uint4*)(&a_lds[nn * 32 + 16 + kofsA]);
        a1.q[0] = *(const uint4*)(&a_lds[(16 + nn) * 32 + kofsA]);
        a1.q[1] = *(const uint4*)(&a_lds[(16 + nn) * 32 + 16 + kofsA]);

        const unsigned short* bbase = w1t + (size_t)kt * 32 + khalfB;
#pragma unroll
        for (int t = 0; t < TPW; ++t) {
            Frag bf;
            const unsigned short* bp = bbase + (size_t)((ct0 + t) * 16 + nn) * KF;
            bf.q[0] = *(const uint4*)bp;
            bf.q[1] = *(const uint4*)(bp + 16);
            acc[0][t] = __builtin_amdgcn_wmma_f32_16x16x32_bf16(
                false, a0.v, false, bf.v, (short)0, acc[0][t], false, false);
            acc[1][t] = __builtin_amdgcn_wmma_f32_16x16x32_bf16(
                false, a1.v, false, bf.v, (short)0, acc[1][t], false, false);
        }
        __syncthreads();
    }

    // --- fused epilogue: +b1, relu, dot with w2, reduce across columns ---
    if (tid < 32) sc[tid] = b2[0];
    __syncthreads();

    const int rbase = hi * 8; // acc VGPR r -> row r (lanes<16) or r+8 (lanes>=16)
#pragma unroll
    for (int q = 0; q < RT; ++q) {
        float part[8];
#pragma unroll
        for (int r = 0; r < 8; ++r) part[r] = 0.f;
#pragma unroll
        for (int t = 0; t < TPW; ++t) {
#pragma unroll
            for (int r = 0; r < 8; ++r) {
                float h = acc[q][t][r] + b1c[t];
                h = fmaxf(h, 0.f);
                part[r] += h * w2c[t];
            }
        }
#pragma unroll
        for (int r = 0; r < 8; ++r) atomicAdd(&sc[q * 16 + rbase + r], part[r]);
    }
    __syncthreads();

    if (tid < 32) {
        int R = rowBase + tid;
        int b = R / LP; int rem = R - b * LP;
        int l = rem / Pn; int p = rem - l * Pn;
        int jc = l + p; if (jc > Ln - 1) jc = Ln - 1;
        int cnt = csm[b * (Ln + 1) + jc + 1] - csm[b * (Ln + 1) + l];
        bool valid = (l + p < Ln) && (cnt == p + 1);
        scores[R] = valid ? sc[tid] : -__builtin_inff();
    }
}

// ---------------------------------------------------------------------------
// K3: per-batch top-32 by iterative argmax in LDS; emits masks + phrase_scores
// ---------------------------------------------------------------------------
__global__ __launch_bounds__(256) void k_topk(const float* __restrict__ scores,
                                              int* __restrict__ top_idx,
                                              float* __restrict__ top_scores,
                                              int* __restrict__ n_valid,
                                              float* __restrict__ out_masks,
                                              float* __restrict__ out_pscores) {
    int b = blockIdx.x, tid = threadIdx.x;
    __shared__ float s[LP];
    __shared__ float rv[256];
    __shared__ int   ri[256];
    __shared__ float ts[Mn];
    __shared__ int   tix[Mn];
    __shared__ int   cnt;
    if (tid == 0) cnt = 0;
    __syncthreads();
    int local = 0;
    for (int i = tid; i < LP; i += 256) {
        float v = scores[(size_t)b * LP + i];
        s[i] = v;
        if (v > -1e37f) local++;
    }
    atomicAdd(&cnt, local);
    __syncthreads();

    for (int m = 0; m < Mn; ++m) {
        float best = -__builtin_inff(); int bi = 0;
        for (int i = tid; i < LP; i += 256) {
            float v = s[i];
            if (v > best) { best = v; bi = i; }
        }
        rv[tid] = best; ri[tid] = bi;
        __syncthreads();
        for (int off = 128; off > 0; off >>= 1) {
            if (tid < off) {
                float o = rv[tid + off]; int oi = ri[tid + off];
                if (o > rv[tid] || (o == rv[tid] && oi < ri[tid])) { rv[tid] = o; ri[tid] = oi; }
            }
            __syncthreads();
        }
        if (tid == 0) { ts[m] = rv[0]; tix[m] = ri[0]; s[ri[0]] = -__builtin_inff(); }
        __syncthreads();
    }

    if (tid < Mn) {
        top_idx[b * Mn + tid] = tix[tid];
        top_scores[b * Mn + tid] = ts[tid];
        int vm = (tid < cnt) ? 1 : 0;
        out_masks[b * Mn + tid] = vm ? 1.0f : 0.0f;
        out_pscores[b * Mn + tid] = vm ? ts[tid] : -__builtin_inff();
    }
    if (tid == 0) n_valid[b] = cnt;
}

// ---------------------------------------------------------------------------
// K4: gather masked mean-features for the top-M spans
// ---------------------------------------------------------------------------
__global__ __launch_bounds__(256) void k_gather(const float* __restrict__ csh,
                                                const int* __restrict__ top_idx,
                                                const int* __restrict__ n_valid,
                                                float* __restrict__ pf) {
    int bm = blockIdx.x;
    int b = bm / Mn; int m = bm - b * Mn;
    int tid = threadIdx.x;
    float* row = pf + (size_t)bm * Hn;
    if (m >= n_valid[b]) {
        for (int c = tid; c < Hn; c += 256) row[c] = 0.f;
        return;
    }
    int idx = top_idx[bm];
    int l = idx / Pn; int p = idx - l * Pn;
    int jc = l + p; if (jc > Ln - 1) jc = Ln - 1;
    const float* c0 = csh + ((size_t)(b * (Ln + 1) + l)) * Hn;
    const float* c1 = csh + ((size_t)(b * (Ln + 1) + jc + 1)) * Hn;
    float il = 1.f / (float)(p + 1);
    for (int c = tid; c < Hn; c += 256) row[c] = (c1[c] - c0[c]) * il;
}

// ---------------------------------------------------------------------------
// K5: gate MLP + softmax attention + output projection (tiny tail, scalar)
// ---------------------------------------------------------------------------
__global__ __launch_bounds__(256) void k_final(const float* __restrict__ pf,
                                               const int* __restrict__ n_valid,
                                               const float* __restrict__ Wg1,
                                               const float* __restrict__ bg1,
                                               const float* __restrict__ wg2,
                                               const float* __restrict__ bg2,
                                               const float* __restrict__ Wp,
                                               const float* __restrict__ bp,
                                               float* __restrict__ out_embeds,
                                               float* __restrict__ out_att) {
    int b = blockIdx.x, tid = threadIdx.x;
    int nv = n_valid[b];
    const float* pfb = pf + (size_t)b * Mn * Hn;
    __shared__ float red[256];
    __shared__ float gate[Mn];

    for (int m = 0; m < Mn; ++m) {
        float part = 0.f;
        const float* prow = pfb + (size_t)m * Hn;
        for (int g = tid; g < Gn; g += 256) {
            float acc = bg1[g];
            for (int ci = 0; ci < Hn; ++ci) acc += prow[ci] * Wg1[(size_t)ci * Gn + g];
            part += tanhf(acc) * wg2[g];
        }
        red[tid] = part;
        __syncthreads();
        for (int off = 128; off > 0; off >>= 1) {
            if (tid < off) red[tid] += red[tid + off];
            __syncthreads();
        }
        if (tid == 0) gate[m] = red[0] + bg2[0];
        __syncthreads();
    }

    if (tid == 0) {
        float mx = -__builtin_inff();
        for (int m = 0; m < Mn; ++m) {
            float g = (m < nv) ? gate[m] : -__builtin_inff();
            gate[m] = g;
            if (g > mx) mx = g;
        }
        float ssum = 0.f;
        float e[Mn];
        for (int m = 0; m < Mn; ++m) {
            float v = (gate[m] > -__builtin_inff()) ? __expf(gate[m] - mx) : 0.f;
            e[m] = v; ssum += v;
        }
        float inv = (ssum > 0.f) ? 1.f / ssum : 0.f;
        for (int m = 0; m < Mn; ++m) out_att[b * Mn + m] = e[m] * inv;
    }

    for (int o = tid; o < Mn * Dn; o += 256) {
        int m = o / Dn; int d = o - m * Dn;
        float acc = bp[d];
        const float* prow = pfb + (size_t)m * Hn;
        for (int ci = 0; ci < Hn; ++ci) acc += prow[ci] * Wp[(size_t)ci * Dn + d];
        out_embeds[((size_t)b * Mn + m) * Dn + d] = acc;
    }
}

// ---------------------------------------------------------------------------
extern "C" void kernel_launch(void* const* d_in, const int* in_sizes, int n_in,
                              void* d_out, int out_size, void* d_ws, size_t ws_size,
                              hipStream_t stream) {
    const float* hidden = (const float*)d_in[0];
    const int*   amask  = (const int*)d_in[1];
    const float* W1     = (const float*)d_in[2];
    const float* b1     = (const float*)d_in[3];
    const float* w2     = (const float*)d_in[4];
    const float* b2     = (const float*)d_in[5];
    const float* Wg1    = (const float*)d_in[6];
    const float* bg1    = (const float*)d_in[7];
    const float* wg2    = (const float*)d_in[8];
    const float* bg2    = (const float*)d_in[9];
    const float* Wp     = (const float*)d_in[10];
    const float* bp     = (const float*)d_in[11];
    (void)in_sizes; (void)n_in; (void)out_size; (void)ws_size;

    float* out = (float*)d_out;
    float* out_embeds  = out;                       // B*M*D = 65536
    float* out_masks   = out_embeds + Bn * Mn * Dn; // B*M   = 256
    float* out_att     = out_masks + Bn * Mn;       // B*M   = 256
    float* out_pscores = out_att + Bn * Mn;         // B*M   = 256

    char* ws = (char*)d_ws;
    size_t off = 0;
    auto al = [](size_t x) { return (x + 255) & ~(size_t)255; };
    float* csh = (float*)(ws + off);            off = al(off + (size_t)Bn * (Ln + 1) * Hn * 4);
    int* csm = (int*)(ws + off);                off = al(off + (size_t)Bn * (Ln + 1) * 4);
    unsigned short* w1t = (unsigned short*)(ws + off); off = al(off + (size_t)Hn * KF * 2);
    float* scoresWS = (float*)(ws + off);       off = al(off + (size_t)Bn * LP * 4);
    int* top_idx = (int*)(ws + off);            off = al(off + (size_t)Bn * Mn * 4);
    float* top_scores = (float*)(ws + off);     off = al(off + (size_t)Bn * Mn * 4);
    int* n_valid = (int*)(ws + off);            off = al(off + (size_t)Bn * 4);
    float* pf = (float*)(ws + off);             off = al(off + (size_t)Bn * Mn * Hn * 4);

    k_cumsum<<<Bn, 256, 0, stream>>>(hidden, amask, csh, csm);
    k_prep_w1<<<(Hn * KF + 255) / 256, 256, 0, stream>>>(W1, w1t);
    k_score<<<ROWS / 32, 256, 0, stream>>>(hidden, csh, csm, w1t, b1, w2, b2, scoresWS);
    k_topk<<<Bn, 256, 0, stream>>>(scoresWS, top_idx, top_scores, n_valid, out_masks, out_pscores);
    k_gather<<<Bn * Mn, 256, 0, stream>>>(csh, top_idx, n_valid, pf);
    k_final<<<Bn, 256, 0, stream>>>(pf, n_valid, Wg1, bg1, wg2, bg2, Wp, bp, out_embeds, out_att);
}
